// LightningAttention_70978629533879
// MI455X (gfx1250) — compile-verified, matrix operand order checked by reading, not run on hardware
//
#include <hip/hip_runtime.h>
#include <hip/hip_bf16.h>

// ---------------------------------------------------------------------------
// Problem constants (match reference)
// ---------------------------------------------------------------------------
#define BSZ 2
#define SEQ 2048
#define DM  2048
#define NH  16
#define KVH 4
#define HD  128
#define CHUNK 128
#define NCHUNK (SEQ / CHUNK)   // 16
#define EPS 1e-6f

typedef __attribute__((ext_vector_type(16))) __bf16 v16bf;
typedef __attribute__((ext_vector_type(8)))  __bf16 v8bf;
typedef __attribute__((ext_vector_type(8)))  float  v8f;
typedef __attribute__((__vector_size__(4 * sizeof(int)))) int v4i;

union FragU { v16bf v; v8bf h[2]; };

// --------------------------- async LDS copy path ---------------------------
#if defined(__has_builtin)
#  if __has_builtin(__builtin_amdgcn_global_load_async_to_lds_b128) && \
      __has_builtin(__builtin_amdgcn_s_wait_asynccnt)
#    define HAVE_ASYNC_LDS 1
#  endif
#endif
#ifndef HAVE_ASYNC_LDS
#  define HAVE_ASYNC_LDS 0
#endif

#define AS1 __attribute__((address_space(1)))
#define AS3 __attribute__((address_space(3)))

// 16-byte global -> LDS copy. Async (ASYNCcnt-tracked) when available.
// Builtin signature (from hipcc diagnostic): (as1 v4i*, as3 v4i*, imm, imm).
static __device__ inline void async_copy16(const __bf16* g, __bf16* l) {
#if HAVE_ASYNC_LDS
  // generic LDS pointer truncates to the 32-bit LDS offset (ISA aperture rule)
  __builtin_amdgcn_global_load_async_to_lds_b128(
      (AS1 v4i*)(unsigned long long)(const void*)g,
      (AS3 v4i*)(unsigned int)(unsigned long long)(const void*)l, 0, 0);
#else
  *(v8bf*)l = *(const v8bf*)g;
#endif
}

static __device__ inline void async_wait() {
#if HAVE_ASYNC_LDS
  __builtin_amdgcn_s_wait_asynccnt(0);
#endif
}

// ----------------------------- bf16 helpers --------------------------------
static __device__ inline __bf16 f2bf(float f) {
  union { float f; unsigned u; } a; a.f = f;
  unsigned u = a.u;
  unsigned r = (u + 0x7FFFu + ((u >> 16) & 1u)) >> 16;
  union { unsigned short s; __bf16 b; } o; o.s = (unsigned short)r;
  return o.b;
}

static __device__ inline unsigned pack2bf(float x, float y) {
  union { float f; unsigned u; } a, b; a.f = x; b.f = y;
  unsigned ra = (a.u + 0x7FFFu + ((a.u >> 16) & 1u)) >> 16;
  unsigned rb = (b.u + 0x7FFFu + ((b.u >> 16) & 1u)) >> 16;
  return (rb << 16) | (ra & 0xFFFFu);
}

static __device__ inline v8f vzero8() {
  v8f v;
  #pragma unroll
  for (int i = 0; i < 8; ++i) v[i] = 0.f;
  return v;
}

// A-style fragment load from a row-major LDS tile.
// CDNA5 16-bit A layout (wave32): lane l (l<16): M=l, K=hi*8+{0..7} in
// VGPRs0-3 and K=16+hi*8+{0..7} in VGPRs4-7, hi = lane>>4.
static __device__ inline v16bf load_frag(const __bf16* base, int stride,
                                         int row0, int k0) {
  int lane = threadIdx.x & 31;
  int l = lane & 15, hi = lane >> 4;
  const __bf16* p = base + (row0 + l) * stride + k0 + hi * 8;
  FragU u;
  u.h[0] = *(const v8bf*)(p);
  u.h[1] = *(const v8bf*)(p + 16);
  return u.v;
}

static __device__ inline v8f wmma_bf16(v16bf a, v16bf b, v8f c) {
  return __builtin_amdgcn_wmma_f32_16x16x32_bf16(
      false, a, false, b, (short)0, c, false, false);
}

// ---------------------------------------------------------------------------
// Streaming f32 -> bf16 conversion (vectorized x8). n multiple of 2048.
// ---------------------------------------------------------------------------
__global__ void conv_bf16_kernel(const float* __restrict__ x,
                                 __bf16* __restrict__ y) {
  size_t i = ((size_t)blockIdx.x * 256 + threadIdx.x) * 8;
  float4 a = *(const float4*)(x + i);
  float4 b = *(const float4*)(x + i + 4);
  unsigned o[4];
  o[0] = pack2bf(a.x, a.y);
  o[1] = pack2bf(a.z, a.w);
  o[2] = pack2bf(b.x, b.y);
  o[3] = pack2bf(b.z, b.w);
  *(uint4*)(y + i) = *(uint4*)o;
}

// ---------------------------------------------------------------------------
// GEMM: Y[M,N] (f32) = X[M,K] (bf16) @ W[N,K]^T (bf16), bf16 WMMA,
// double-buffered async global->LDS staging. Block 256 (8 waves);
// wave w owns rows m0+16w..+15 x 128 cols (8 accum frags).
// ---------------------------------------------------------------------------
__global__ void gemm_bf16_kernel(const __bf16* __restrict__ X,
                                 const __bf16* __restrict__ W,
                                 float* __restrict__ Y,
                                 int M, int N, int K) {
  __shared__ __bf16 Xs[2 * 128 * 40];   // stride 40: bank-conflict-free b128
  __shared__ __bf16 Ws[2 * 128 * 40];

  const int tid = threadIdx.x;
  const int w = tid >> 5, lane = tid & 31, l = lane & 15, hi = lane >> 4;
  const int m0 = blockIdx.y * 128, n0 = blockIdx.x * 128;

  v8f acc[8];
  #pragma unroll
  for (int nb = 0; nb < 8; ++nb) acc[nb] = vzero8();

  auto stage = [&](int k0, int buf) {
    __bf16* xd = Xs + buf * (128 * 40);
    __bf16* wd = Ws + buf * (128 * 40);
    #pragma unroll
    for (int j = 0; j < 2; ++j) {
      int id = j * 256 + tid;           // 512 16B-chunks per matrix
      int row = id >> 2, cc = (id & 3) * 8;
      async_copy16(X + (size_t)(m0 + row) * K + k0 + cc, xd + row * 40 + cc);
      async_copy16(W + (size_t)(n0 + row) * K + k0 + cc, wd + row * 40 + cc);
    }
  };

  int cur = 0;
  stage(0, cur);
  async_wait();
  __syncthreads();

  for (int k0 = 0; k0 < K; k0 += 32) {
    int nxt = cur ^ 1;
    if (k0 + 32 < K) stage(k0 + 32, nxt);   // prefetch while computing

    const __bf16* xb = Xs + cur * (128 * 40);
    const __bf16* wb = Ws + cur * (128 * 40);
    v16bf a = load_frag(xb, 40, 16 * w, 0);
    v16bf bb[8];
    #pragma unroll
    for (int nb = 0; nb < 8; ++nb) bb[nb] = load_frag(wb, 40, 16 * nb, 0);
    #pragma unroll
    for (int nb = 0; nb < 8; ++nb) acc[nb] = wmma_bf16(a, bb[nb], acc[nb]);

    async_wait();
    __syncthreads();
    cur = nxt;
  }

  // C/D layout: VGPR r holds M = r + 8*hi, lane l holds N = l.
  #pragma unroll
  for (int nb = 0; nb < 8; ++nb) {
    #pragma unroll
    for (int r = 0; r < 8; ++r) {
      int row = m0 + 16 * w + r + 8 * hi;
      int col = n0 + 16 * nb + l;
      Y[(size_t)row * N + col] = acc[nb][r];
    }
  }
}

// ---------------------------------------------------------------------------
// RoPE + elu(x)+1 feature map for Q. qp: [B,S,H*HD] f32 -> Qn: [B,H,S,HD] bf16.
// One thread handles the (d, d+64) rotation pair.
// ---------------------------------------------------------------------------
__global__ void rope_feat_q_kernel(const float* __restrict__ qp,
                                   __bf16* __restrict__ Qn) {
  int t = blockIdx.x * 256 + threadIdx.x;        // B*S*H*64 threads
  int d = t & 63; t >>= 6;
  int h = t & (NH - 1); t >>= 4;
  int s = t & (SEQ - 1);
  int b = t >> 11;
  float inv = __powf(10000.f, -(2.f * (float)d) / 128.f);
  float th = (float)s * inv;
  float sn, cs;
  __sincosf(th, &sn, &cs);
  size_t base = ((size_t)b * SEQ + s) * DM + h * HD + d;
  float x1 = qp[base], x2 = qp[base + 64];
  float r1 = x1 * cs - x2 * sn;
  float r2 = x2 * cs + x1 * sn;
  r1 = (r1 > 0.f) ? (r1 + 1.f) : __expf(r1);     // elu(x)+1
  r2 = (r2 > 0.f) ? (r2 + 1.f) : __expf(r2);
  size_t ob = (((size_t)b * NH + h) * SEQ + s) * HD + d;
  Qn[ob] = f2bf(r1);
  Qn[ob + 64] = f2bf(r2);
}

// Same for K, writing both normal [B,KVH,S,HD] and transposed [B,KVH,HD,S].
__global__ void rope_feat_k_kernel(const float* __restrict__ kp,
                                   __bf16* __restrict__ Kn,
                                   __bf16* __restrict__ Kt) {
  int t = blockIdx.x * 256 + threadIdx.x;        // B*S*KVH*64 threads
  int d = t & 63; t >>= 6;
  int kh = t & (KVH - 1); t >>= 2;
  int s = t & (SEQ - 1);
  int b = t >> 11;
  float inv = __powf(10000.f, -(2.f * (float)d) / 128.f);
  float th = (float)s * inv;
  float sn, cs;
  __sincosf(th, &sn, &cs);
  size_t base = ((size_t)b * SEQ + s) * (KVH * HD) + kh * HD + d;
  float x1 = kp[base], x2 = kp[base + 64];
  float r1 = x1 * cs - x2 * sn;
  float r2 = x2 * cs + x1 * sn;
  r1 = (r1 > 0.f) ? (r1 + 1.f) : __expf(r1);
  r2 = (r2 > 0.f) ? (r2 + 1.f) : __expf(r2);
  __bf16 b1 = f2bf(r1), b2 = f2bf(r2);
  size_t on = (((size_t)b * KVH + kh) * SEQ + s) * HD + d;
  Kn[on] = b1;
  Kn[on + 64] = b2;
  size_t ot = (((size_t)b * KVH + kh) * HD + d) * SEQ + s;
  Kt[ot] = b1;
  Kt[ot + (size_t)64 * SEQ] = b2;
}

// V: convert f32 -> bf16, transposed layout [B,KVH,HD,S].
__global__ void v_transpose_kernel(const float* __restrict__ vp,
                                   __bf16* __restrict__ Vt) {
  int t = blockIdx.x * 256 + threadIdx.x;        // B*S*KVH*128 threads
  int d = t & 127; t >>= 7;
  int kh = t & (KVH - 1); t >>= 2;
  int s = t & (SEQ - 1);
  int b = t >> 11;
  float x = vp[((size_t)b * SEQ + s) * (KVH * HD) + kh * HD + d];
  Vt[(((size_t)b * KVH + kh) * HD + d) * SEQ + s] = f2bf(x);
}

// ---------------------------------------------------------------------------
// Chunked causal linear attention. One block per (b,h); 256 threads = 8 waves.
// Sequential over 16 chunks of 128. State[128,128] f32 kept in LDS in
// transposed (StT[n_out][k_feat]) layout + bf16 mirror for the Q@State WMMA.
// LDS strides padded: 136 (bf16), 132 (f32). Output written as bf16 (feeds
// the bf16 Wo GEMM).
// ---------------------------------------------------------------------------
#define SB 136
#define SF 132
#define SMEM_ATTN (6 * 128 * SB * 2 + 128 * SF * 4 + 3 * 128 * 4)

__global__ void lin_attn_kernel(const __bf16* __restrict__ Qn,
                                const __bf16* __restrict__ Kn,
                                const __bf16* __restrict__ Kt,
                                const __bf16* __restrict__ Vt,
                                __bf16* __restrict__ Oattn) {
  extern __shared__ char smem[];
  __bf16* Qs  = (__bf16*)smem;              // [128][SB] query (time x feat)
  __bf16* Kns = Qs  + 128 * SB;             // key, normal (time x feat)
  __bf16* Kts = Kns + 128 * SB;             // key, transposed (feat x time)
  __bf16* Vts = Kts + 128 * SB;             // value, transposed (dout x time)
  __bf16* Ps  = Vts + 128 * SB;             // masked scores (tq x tk)
  __bf16* Stb = Ps  + 128 * SB;             // bf16 state mirror StT[n][k]
  float*  StF = (float*)(Stb + 128 * SB);   // f32 state StT[n][k], stride SF
  float*  zF  = StF + 128 * SF;             // z accumulator [feat]
  float*  den = zF + 128;                   // intra-chunk denominator
  float*  qz  = den + 128;                  // q . z_prev

  const int tid = threadIdx.x;
  const int w = tid >> 5, lane = tid & 31, l = lane & 15, hi = lane >> 4;
  const int bh = blockIdx.x;
  const int h = bh % NH, b = bh / NH, kh = h >> 2;   // GQA: REP = 4

  for (int i = tid; i < 128 * SF; i += 256) StF[i] = 0.f;
  if (tid < 128) zF[tid] = 0.f;
  __syncthreads();

  const __bf16* Qg  = Qn + ((size_t)(b * NH + h) * SEQ) * HD;
  const __bf16* Kg  = Kn + ((size_t)(b * KVH + kh) * SEQ) * HD;
  const __bf16* Ktg = Kt + ((size_t)(b * KVH + kh) * HD) * SEQ;
  const __bf16* Vtg = Vt + ((size_t)(b * KVH + kh) * HD) * SEQ;

  for (int c = 0; c < NCHUNK; ++c) {
    const int t0 = c * CHUNK;

    // ---- Phase A: async-stage tiles; convert state snapshot meanwhile -----
    #pragma unroll
    for (int j = 0; j < 8; ++j) {
      int idx = j * 256 + tid;              // 2048 16B-chunks per tile
      int row = idx >> 4, cc = (idx & 15) * 8;
      async_copy16(Qg  + (size_t)(t0 + row) * HD + cc, Qs  + row * SB + cc);
      async_copy16(Kg  + (size_t)(t0 + row) * HD + cc, Kns + row * SB + cc);
      async_copy16(Ktg + (size_t)row * SEQ + t0 + cc,  Kts + row * SB + cc);
      async_copy16(Vtg + (size_t)row * SEQ + t0 + cc,  Vts + row * SB + cc);
    }
    // f32 state -> bf16 mirror (overlaps with async copies in flight)
    for (int j = 0; j < 32; ++j) {
      int p = j * 256 + tid;                // 8192 element-pairs
      int row = p >> 6, col = (p & 63) * 2;
      *(unsigned*)&Stb[row * SB + col] =
          pack2bf(StF[row * SF + col], StF[row * SF + col + 1]);
    }
    async_wait();
    __syncthreads();

    // q . z_prev (before z is updated this chunk)
    if (tid < 128) {
      float a = 0.f;
      for (int d0 = 0; d0 < 128; ++d0)
        a += (float)Qs[tid * SB + d0] * zF[d0];
      qz[tid] = a;
    }

    // ---- Phase B: S_intra = Q K^T, causal mask, row-sum, Ps ---------------
    float rs[8];
    #pragma unroll
    for (int r = 0; r < 8; ++r) rs[r] = 0.f;

    for (int nb = 0; nb < 8; ++nb) {
      if (nb <= w) {
        v8f sacc = vzero8();
        #pragma unroll
        for (int kk = 0; kk < 4; ++kk) {
          v16bf a = load_frag(Qs, SB, 16 * w, 32 * kk);
          v16bf bb = load_frag(Kns, SB, 16 * nb, 32 * kk);
          sacc = wmma_bf16(a, bb, sacc);
        }
        #pragma unroll
        for (int r = 0; r < 8; ++r) {
          float v = sacc[r];
          if (nb == w && l > r + 8 * hi) v = 0.f;   // causal: keep j <= i
          rs[r] += v;
          Ps[(16 * w + r + 8 * hi) * SB + 16 * nb + l] = f2bf(v);
        }
      } else {
        #pragma unroll
        for (int r = 0; r < 8; ++r)
          Ps[(16 * w + r + 8 * hi) * SB + 16 * nb + l] = f2bf(0.f);
      }
    }
    // reduce masked row-sums across the 16-lane half (xor <= 8 stays in-half)
    #pragma unroll
    for (int r = 0; r < 8; ++r) {
      float v = rs[r];
      v += __shfl_xor(v, 1, 32);
      v += __shfl_xor(v, 2, 32);
      v += __shfl_xor(v, 4, 32);
      v += __shfl_xor(v, 8, 32);
      if (l == 0) den[16 * w + r + 8 * hi] = v;
    }
    __syncthreads();

    // ---- Phase C: O = mask(S) @ V + Q @ State_prev; normalize; store ------
    v8f acc[8];
    #pragma unroll
    for (int nb = 0; nb < 8; ++nb) acc[nb] = vzero8();

    const int kkmax = (w >> 1) + 1;   // only non-zero causal K-blocks
    for (int kk = 0; kk < kkmax; ++kk) {
      v16bf a = load_frag(Ps, SB, 16 * w, 32 * kk);
      v16bf bb[8];
      #pragma unroll
      for (int nb = 0; nb < 8; ++nb) bb[nb] = load_frag(Vts, SB, 16 * nb, 32 * kk);
      #pragma unroll
      for (int nb = 0; nb < 8; ++nb) acc[nb] = wmma_bf16(a, bb[nb], acc[nb]);
    }
    #pragma unroll
    for (int kk = 0; kk < 4; ++kk) {
      v16bf a = load_frag(Qs, SB, 16 * w, 32 * kk);
      v16bf bb[8];
      #pragma unroll
      for (int nb = 0; nb < 8; ++nb) bb[nb] = load_frag(Stb, SB, 16 * nb, 32 * kk);
      #pragma unroll
      for (int nb = 0; nb < 8; ++nb) acc[nb] = wmma_bf16(a, bb[nb], acc[nb]);
    }
    #pragma unroll
    for (int nb = 0; nb < 8; ++nb) {
      #pragma unroll
      for (int r = 0; r < 8; ++r) {
        int row = 16 * w + r + 8 * hi;
        float dd = den[row] + qz[row] + EPS;
        float val = acc[nb][r] / dd;
        size_t srow = (size_t)b * SEQ + t0 + row;
        Oattn[srow * (NH * HD) + h * HD + 16 * nb + l] = f2bf(val);
      }
    }
    __syncthreads();

    // ---- Phase D: State += K^T @ V ; z += sum_t k ------------------------
    v8f cf[8];
    #pragma unroll
    for (int nb = 0; nb < 8; ++nb) {
      float* p = &StF[(16 * nb + l) * SF + 16 * w + 8 * hi];
      #pragma unroll
      for (int r = 0; r < 8; ++r) cf[nb][r] = p[r];
    }
    #pragma unroll
    for (int kk = 0; kk < 4; ++kk) {
      v16bf a = load_frag(Kts, SB, 16 * w, 32 * kk);
      v16bf bb[8];
      #pragma unroll
      for (int nb = 0; nb < 8; ++nb) bb[nb] = load_frag(Vts, SB, 16 * nb, 32 * kk);
      #pragma unroll
      for (int nb = 0; nb < 8; ++nb) cf[nb] = wmma_bf16(a, bb[nb], cf[nb]);
    }
    #pragma unroll
    for (int nb = 0; nb < 8; ++nb) {
      float* p = &StF[(16 * nb + l) * SF + 16 * w + 8 * hi];
      #pragma unroll
      for (int r = 0; r < 8; ++r) p[r] = cf[nb][r];
    }
    if (tid < 128) {
      float a = 0.f;
      for (int j = 0; j < 128; ++j) a += (float)Kts[tid * SB + j];
      zF[tid] += a;
    }
    __syncthreads();
  }
}

// ---------------------------------------------------------------------------
// Host-side launch
// ---------------------------------------------------------------------------
extern "C" void kernel_launch(void* const* d_in, const int* in_sizes, int n_in,
                              void* d_out, int out_size, void* d_ws, size_t ws_size,
                              hipStream_t stream) {
  (void)in_sizes; (void)n_in; (void)out_size; (void)ws_size;
  const float* x  = (const float*)d_in[0];
  const float* Wq = (const float*)d_in[1];
  const float* Wk = (const float*)d_in[2];
  const float* Wv = (const float*)d_in[3];
  const float* Wo = (const float*)d_in[4];
  float* out = (float*)d_out;

  char* ws = (char*)d_ws;
  size_t off = 0;
  const size_t M = (size_t)BSZ * SEQ;                  // 4096
  __bf16* xb  = (__bf16*)(ws + off); off += M * DM * 2;            // 16 MB
  __bf16* Wqb = (__bf16*)(ws + off); off += (size_t)DM * DM * 2;   //  8 MB
  __bf16* Wkb = (__bf16*)(ws + off); off += (size_t)KVH * HD * DM * 2;
  __bf16* Wvb = (__bf16*)(ws + off); off += (size_t)KVH * HD * DM * 2;
  __bf16* Wob = (__bf16*)(ws + off); off += (size_t)DM * DM * 2;   //  8 MB
  float* qproj = (float*)(ws + off); off += M * DM * 4;            // 32 MB
  float* kproj = (float*)(ws + off); off += M * (KVH * HD) * 4;    //  8 MB
  float* vproj = (float*)(ws + off); off += M * (KVH * HD) * 4;    //  8 MB
  __bf16* Qn = (__bf16*)(ws + off); off += M * DM * 2;             // 16 MB
  __bf16* Kn = (__bf16*)(ws + off); off += M * (KVH * HD) * 2;     //  4 MB
  __bf16* Kt = (__bf16*)(ws + off); off += M * (KVH * HD) * 2;     //  4 MB
  __bf16* Vt = (__bf16*)(ws + off); off += M * (KVH * HD) * 2;     //  4 MB
  __bf16* OattnB = (__bf16*)qproj;  // reuse: qproj fully consumed first

  // 0) One-time bf16 conversions (halves GEMM read bandwidth; enables raw
  //    async global->LDS staging with no conversion in the copy path)
  conv_bf16_kernel<<<(M * DM) / 2048, 256, 0, stream>>>(x, xb);
  conv_bf16_kernel<<<((size_t)DM * DM) / 2048, 256, 0, stream>>>(Wq, Wqb);
  conv_bf16_kernel<<<((size_t)KVH * HD * DM) / 2048, 256, 0, stream>>>(Wk, Wkb);
  conv_bf16_kernel<<<((size_t)KVH * HD * DM) / 2048, 256, 0, stream>>>(Wv, Wvb);
  conv_bf16_kernel<<<((size_t)DM * DM) / 2048, 256, 0, stream>>>(Wo, Wob);

  // 1) Projections (bf16 WMMA GEMMs, async double-buffered staging)
  gemm_bf16_kernel<<<dim3(DM / 128, M / 128), 256, 0, stream>>>(
      xb, Wqb, qproj, (int)M, DM, DM);
  gemm_bf16_kernel<<<dim3((KVH * HD) / 128, M / 128), 256, 0, stream>>>(
      xb, Wkb, kproj, (int)M, KVH * HD, DM);
  gemm_bf16_kernel<<<dim3((KVH * HD) / 128, M / 128), 256, 0, stream>>>(
      xb, Wvb, vproj, (int)M, KVH * HD, DM);

  // 2) RoPE + elu+1 feature map, layout shuffles to bf16
  rope_feat_q_kernel<<<(BSZ * SEQ * NH * 64) / 256, 256, 0, stream>>>(qproj, Qn);
  rope_feat_k_kernel<<<(BSZ * SEQ * KVH * 64) / 256, 256, 0, stream>>>(kproj, Kn, Kt);
  v_transpose_kernel<<<(BSZ * SEQ * KVH * HD) / 256, 256, 0, stream>>>(vproj, Vt);

  // 3) Chunked causal linear attention (one block per (b,h))
  (void)hipFuncSetAttribute(reinterpret_cast<const void*>(lin_attn_kernel),
                            hipFuncAttributeMaxDynamicSharedMemorySize, SMEM_ATTN);
  lin_attn_kernel<<<BSZ * NH, 256, SMEM_ATTN, stream>>>(Qn, Kn, Kt, Vt, OattnB);

  // 4) Output projection
  gemm_bf16_kernel<<<dim3(DM / 128, M / 128), 256, 0, stream>>>(
      OattnB, Wob, out, (int)M, DM, DM);
}
